// ODE_RNN_10350871183741
// MI455X (gfx1250) — matrix operand order repaired and checked by reading, not compiled, not run on hardware
//
#include <hip/hip_runtime.h>

// ---------------------------------------------------------------------------
// ODE-RNN persistent kernel for gfx1250 (MI455X).
// Grid: 8 blocks x 256 threads (8 wave32 waves). Block b owns samples
// [16b, 16b+16). Whole T=64 sequence computed in one launch; dynamics MLP
// weights live in LDS as f16; matmuls via v_wmma_f32_16x16x32_f16.
// ---------------------------------------------------------------------------

#define B_   128
#define T_   64
#define OB_  32
#define AC_  8
#define L_   128
#define H_   256
#define BDIM 256

typedef __attribute__((ext_vector_type(16))) _Float16 v16h;
typedef __attribute__((ext_vector_type(8)))  _Float16 v8h;
typedef __attribute__((ext_vector_type(8)))  float    v8f;

union V16H { v16h v; v8h h8[2]; _Float16 e[16]; };
union V8F  { v8f  v; float e[8]; };
union V8H  { v8h  v; _Float16 e[8]; };

// --- LDS layout (bytes) -----------------------------------------------------
// sWd0 f16 [256][128]   :      0 ..  65536
// sWd1 f16 [256][256]   :  65536 .. 196608
// sWd2 f16 [128][256]   : 196608 .. 262144
// sK   f32 [5][16][128] : 262144 .. 303104   (reused as GRU gate scratch)
// sY   f32 [16][128]    : 303104 .. 311296
// sYS  f16 [16][128]    : 311296 .. 315392
// sAct f16 [16][256]    : 315392 .. 323584
// sH   f32 [16]         : 323584 .. 323648
#define SMEM_BYTES 323648

__device__ __forceinline__ v8f wmma32(v16h a, v16h b, v8f c) {
  return __builtin_amdgcn_wmma_f32_16x16x32_f16(false, a, false, b,
                                                (short)0, c, false, false);
}

// 8 contiguous f32 from LDS (two ds_load_b128).
__device__ __forceinline__ v8f ld8_f32(const float* p) {
  V8F r;
  *(float4*)&r.e[0] = *(const float4*)p;
  *(float4*)&r.e[4] = *(const float4*)(p + 4);
  return r.v;
}

// Convert 8 f32 -> 8 f16 and store as one 16B LDS write.
__device__ __forceinline__ void st8_f16(_Float16* p, const float* e) {
  V8H o;
#pragma unroll
  for (int j = 0; j < 8; ++j) o.e[j] = (_Float16)e[j];
  *(v8h*)p = o.v;
}

// A fragment (16xK f16 row-major in LDS), K-chunk at kBase.
__device__ __forceinline__ v16h load_a_lds(const _Float16* act, int ld,
                                           int kBase, int lane) {
  const int m = lane & 15, lg = lane >> 4;
  const _Float16* p = act + m * ld + kBase + lg * 16;
  V16H a;
  a.h8[0] = *(const v8h*)(p);        // K = kBase + 16*lg + [0..7]
  a.h8[1] = *(const v8h*)(p + 8);    // K = kBase + 16*lg + 8 + [0..7]
  return a.v;
}

// B fragment from f16 LDS weights W[N][K] row-major (B[k][n] = W[n][k]).
__device__ __forceinline__ v16h load_b_lds(const _Float16* W, int ld,
                                           int nBase, int kBase, int lane) {
  const int n = lane & 15, lg = lane >> 4;
  const _Float16* p = W + (nBase + n) * ld + kBase + lg * 16;
  V16H b;
  b.h8[0] = *(const v8h*)(p);
  b.h8[1] = *(const v8h*)(p + 8);
  return b.v;
}

// B fragment from global f32 weights W[N][K] row-major, converted to f16.
__device__ __forceinline__ v16h load_b_gf32(const float* W, int ld,
                                            int nBase, int kBase, int lane) {
  const int n = lane & 15, lg = lane >> 4;
  const float* p = W + (size_t)(nBase + n) * ld + kBase + lg * 16;
  V16H b;
#pragma unroll
  for (int j = 0; j < 16; ++j) b.e[j] = (_Float16)p[j];
  return b.v;
}

// B fragment from global f32 with K clamp (zero pad), for Wih (K=8).
__device__ __forceinline__ v16h load_b_gf32_kpad(const float* W, int ld,
                                                 int nBase, int lane, int kMax) {
  const int n = lane & 15, lg = lane >> 4;
  V16H b;
#pragma unroll
  for (int j = 0; j < 16; ++j) {
    const int k = lg * 16 + j;
    b.e[j] = (k < kMax) ? (_Float16)W[(size_t)(nBase + n) * ld + k]
                        : (_Float16)0.f;
  }
  return b.v;
}

// A fragment from global f32 activations (rows row0..row0+15, ld floats).
__device__ __forceinline__ v16h load_a_gf32(const float* A, int ld, int row0,
                                            int kBase, int lane) {
  const int m = lane & 15, lg = lane >> 4;
  const float* p = A + (size_t)(row0 + m) * ld + kBase + lg * 16;
  V16H a;
#pragma unroll
  for (int j = 0; j < 8; ++j) a.e[j] = (_Float16)p[j];
#pragma unroll
  for (int j = 0; j < 8; ++j) a.e[8 + j] = (_Float16)p[8 + j];
  return a.v;
}

// A fragment for acs column t (K=8 padded to 32).
__device__ __forceinline__ v16h load_a_acs(const float* acs, int m0, int t,
                                           int lane) {
  const int m = lane & 15, lg = lane >> 4;
  const float* row = acs + (size_t)(m0 + m) * T_ * AC_ + (size_t)t * AC_;
  V16H a;
#pragma unroll
  for (int j = 0; j < 16; ++j) {
    const int k = lg * 16 + j;
    a.e[j] = (k < AC_) ? (_Float16)row[k] : (_Float16)0.f;
  }
  return a.v;
}

__device__ __forceinline__ void store_tile_f16(_Float16* out, int ld, int nBase,
                                               v8f acc, float bias, bool relu,
                                               int lane) {
  const int n = lane & 15, lg = lane >> 4;
  V8F a; a.v = acc;
#pragma unroll
  for (int r = 0; r < 8; ++r) {
    float v = a.e[r] + bias;
    if (relu) v = v > 0.f ? v : 0.f;
    out[(r + 8 * lg) * ld + nBase + n] = (_Float16)v;
  }
}

__device__ __forceinline__ void store_tile_f32(float* out, int ld, int nBase,
                                               v8f acc, float bias, int lane) {
  const int n = lane & 15, lg = lane >> 4;
  V8F a; a.v = acc;
#pragma unroll
  for (int r = 0; r < 8; ++r)
    out[(r + 8 * lg) * ld + nBase + n] = a.e[r] + bias;
}

__device__ __forceinline__ float sigmoidf_(float x) {
  return 1.f / (1.f + __expf(-x));
}

// Dynamics MLP f(y): relu(y@Wd0^T+bd0) -> relu(@Wd1^T+bd1) -> @Wd2^T (no bias;
// caller adds bd2). Input sYS (f16 16x128), scratch sAct (f16 16x256).
// Returns the stage-C 16x16 accumulator for this wave's tile (cols 16w..).
__device__ __forceinline__ v8f feval_core(int w, int lane, const _Float16* sYS,
                                          _Float16* sAct, const _Float16* sWd0,
                                          const _Float16* sWd1,
                                          const _Float16* sWd2, float bd0c0,
                                          float bd0c1, float bd1c0,
                                          float bd1c1) {
  // stage A: h1 = relu(y @ Wd0^T + bd0)   M16 K128 N256
  v8f a0 = {}, a1 = {};
#pragma unroll
  for (int kb = 0; kb < L_; kb += 32) {
    const v16h af = load_a_lds(sYS, L_, kb, lane);
    a0 = wmma32(af, load_b_lds(sWd0, L_, 16 * w, kb, lane), a0);
    a1 = wmma32(af, load_b_lds(sWd0, L_, 16 * (w + 8), kb, lane), a1);
  }
  store_tile_f16(sAct, H_, 16 * w, a0, bd0c0, true, lane);
  store_tile_f16(sAct, H_, 16 * (w + 8), a1, bd0c1, true, lane);
  __syncthreads();
  // stage B: h2 = relu(h1 @ Wd1^T + bd1)  M16 K256 N256 (in-place buffer)
  v8f b0 = {}, b1 = {};
#pragma unroll
  for (int kb = 0; kb < H_; kb += 32) {
    const v16h af = load_a_lds(sAct, H_, kb, lane);
    b0 = wmma32(af, load_b_lds(sWd1, H_, 16 * w, kb, lane), b0);
    b1 = wmma32(af, load_b_lds(sWd1, H_, 16 * (w + 8), kb, lane), b1);
  }
  __syncthreads();  // all reads of h1 done before overwrite
  store_tile_f16(sAct, H_, 16 * w, b0, bd1c0, true, lane);
  store_tile_f16(sAct, H_, 16 * (w + 8), b1, bd1c1, true, lane);
  __syncthreads();
  // stage C: k = h2 @ Wd2^T               M16 K256 N128
  v8f c = {};
#pragma unroll
  for (int kb = 0; kb < H_; kb += 32)
    c = wmma32(load_a_lds(sAct, H_, kb, lane),
               load_b_lds(sWd2, H_, 16 * w, kb, lane), c);
  return c;
}

// Build sYS = f16( y + h * sum_j coef[j]*k_j ), 16x128 elementwise.
// 2048 elems = 256 threads x 8 contiguous -> vector LDS ops.
template <int NC>
__device__ __forceinline__ void build_stage_input(int tid, const float* sY,
                                                  const float* sK,
                                                  const float* sH,
                                                  _Float16* sYS,
                                                  const float* coef) {
  const int i0 = tid * 8;
  const int m = i0 >> 7;
  const float hh = sH[m];
  V8F y; y.v = ld8_f32(sY + i0);
  V8F acc; acc.v = v8f{};
#pragma unroll
  for (int j = 0; j < NC; ++j) {
    V8F kj; kj.v = ld8_f32(sK + j * (16 * L_) + i0);
#pragma unroll
    for (int e = 0; e < 8; ++e) acc.e[e] += coef[j] * kj.e[e];
  }
  float outp[8];
#pragma unroll
  for (int e = 0; e < 8; ++e) outp[e] = y.e[e] + hh * acc.e[e];
  st8_f16(sYS + i0, outp);
  __syncthreads();
}

__launch_bounds__(BDIM, 1)
__global__ void ode_rnn_kernel(
    const float* __restrict__ ob, const float* __restrict__ acs,
    const float* __restrict__ times, const float* __restrict__ We0,
    const float* __restrict__ be0, const float* __restrict__ We1,
    const float* __restrict__ be1, const float* __restrict__ Wd0,
    const float* __restrict__ bd0, const float* __restrict__ Wd1,
    const float* __restrict__ bd1, const float* __restrict__ Wd2,
    const float* __restrict__ bd2, const float* __restrict__ Wo0,
    const float* __restrict__ bo0, const float* __restrict__ Wo1,
    const float* __restrict__ bo1, const float* __restrict__ Wih,
    const float* __restrict__ Whh, const float* __restrict__ bih,
    const float* __restrict__ bn, float* __restrict__ out) {
  __shared__ __align__(16) unsigned char smem[SMEM_BYTES];
  _Float16* sWd0 = (_Float16*)(smem);
  _Float16* sWd1 = (_Float16*)(smem + 65536);
  _Float16* sWd2 = (_Float16*)(smem + 196608);
  float*    sK   = (float*)(smem + 262144);   // 5 x 2048 f32
  float*    sY   = (float*)(smem + 303104);   // 16 x 128 f32
  _Float16* sYS  = (_Float16*)(smem + 311296);
  _Float16* sAct = (_Float16*)(smem + 315392);
  float*    sH   = (float*)(smem + 323584);

  const int tid = threadIdx.x;
  const int w = tid >> 5;       // wave 0..7
  const int lane = tid & 31;
  const int m0 = blockIdx.x * 16;  // first sample of this block
  const int nc0 = 16 * w + (lane & 15);        // this lane's col in tile w
  const int nc1 = 16 * (w + 8) + (lane & 15);  // ... in tile w+8

  // ---- preload dynamics weights into LDS as f16 (vectorized) ---------------
  for (int i = tid * 8; i < H_ * L_; i += BDIM * 8) {
    V8F a; a.v = ld8_f32(Wd0 + i);  // global b128 pair
    st8_f16(sWd0 + i, a.e);
  }
  for (int i = tid * 8; i < H_ * H_; i += BDIM * 8) {
    V8F a; a.v = ld8_f32(Wd1 + i);
    st8_f16(sWd1 + i, a.e);
  }
  for (int i = tid * 8; i < L_ * H_; i += BDIM * 8) {
    V8F a; a.v = ld8_f32(Wd2 + i);
    st8_f16(sWd2 + i, a.e);
  }

  // warm L2/WGP$ for cold weights streamed from global every timestep
  for (int off = tid * 64; off < 384 * 128; off += BDIM * 64)
    __builtin_prefetch(Whh + off, 0, 1);
  for (int off = tid * 64; off < 256 * 128; off += BDIM * 64)
    __builtin_prefetch(Wo0 + off, 0, 1);
  for (int off = tid * 64; off < 32 * 256; off += BDIM * 64)
    __builtin_prefetch(Wo1 + off, 0, 1);

  // per-wave register-cached biases (column-indexed, constant per lane)
  const float bd0c0 = bd0[nc0], bd0c1 = bd0[nc1];
  const float bd1c0 = bd1[nc0], bd1c1 = bd1[nc1];
  const float bd2c0 = bd2[nc0];
  const float be0c0 = be0[nc0], be0c1 = be0[nc1];
  const float be1c0 = be1[nc0];
  const float bo0c0 = bo0[nc0], bo0c1 = bo0[nc1];
  const float bihR = bih[nc0], bihZ = bih[L_ + nc0], bihN = bih[2 * L_ + nc0];
  __syncthreads();

  // ---- encoder: latent0 = relu(ob@We0^T+be0) @ We1^T + be1 -----------------
  {
    const v16h af = load_a_gf32(ob, OB_, m0, 0, lane);  // K=32, one chunk
    v8f e0 = {}, e1 = {};
    e0 = wmma32(af, load_b_gf32(We0, OB_, 16 * w, 0, lane), e0);
    e1 = wmma32(af, load_b_gf32(We0, OB_, 16 * (w + 8), 0, lane), e1);
    store_tile_f16(sAct, H_, 16 * w, e0, be0c0, true, lane);
    store_tile_f16(sAct, H_, 16 * (w + 8), e1, be0c1, true, lane);
  }
  __syncthreads();
  {
    v8f acc = {};
#pragma unroll
    for (int kb = 0; kb < H_; kb += 32)
      acc = wmma32(load_a_lds(sAct, H_, kb, lane),
                   load_b_gf32(We1, H_, 16 * w, kb, lane), acc);
    store_tile_f32(sY, L_, 16 * w, acc, be1c0, lane);
    store_tile_f16(sYS, L_, 16 * w, acc, be1c0, false, lane);
  }
  __syncthreads();

  // Dopri5 tableau
  const float c2[1] = {0.2f};
  const float c3[2] = {3.f / 40.f, 9.f / 40.f};
  const float c4[3] = {44.f / 45.f, -56.f / 15.f, 32.f / 9.f};
  const float c5[4] = {19372.f / 6561.f, -25360.f / 2187.f, 64448.f / 6561.f,
                       -212.f / 729.f};
  const float c6[5] = {9017.f / 3168.f, -355.f / 33.f, 46732.f / 5247.f,
                       49.f / 176.f, -5103.f / 18656.f};
  const float B1 = 35.f / 384.f, B3 = 500.f / 1113.f, B4 = 125.f / 192.f,
              B5 = -2187.f / 6784.f, B6 = 11.f / 84.f;

  for (int t = 0; t < T_; ++t) {
    // ---- integrate latent ODE over [times[t-1], times[t]] (t>0) ------------
    if (t > 0) {
      if (tid < 16) {
        const float* tr = times + (size_t)(m0 + tid) * T_;
        sH[tid] = (tr[t] - tr[t - 1]) * 0.25f;  // dt / N_SUB
      }
      __syncthreads();
      for (int sub = 0; sub < 4; ++sub) {
        // k1 (sYS already == f16(sY))
        v8f k = feval_core(w, lane, sYS, sAct, sWd0, sWd1, sWd2, bd0c0, bd0c1,
                           bd1c0, bd1c1);
        store_tile_f32(sK + 0 * 2048, L_, 16 * w, k, bd2c0, lane);
        __syncthreads();
        build_stage_input<1>(tid, sY, sK, sH, sYS, c2);
        k = feval_core(w, lane, sYS, sAct, sWd0, sWd1, sWd2, bd0c0, bd0c1,
                       bd1c0, bd1c1);
        store_tile_f32(sK + 1 * 2048, L_, 16 * w, k, bd2c0, lane);
        __syncthreads();
        build_stage_input<2>(tid, sY, sK, sH, sYS, c3);
        k = feval_core(w, lane, sYS, sAct, sWd0, sWd1, sWd2, bd0c0, bd0c1,
                       bd1c0, bd1c1);
        store_tile_f32(sK + 2 * 2048, L_, 16 * w, k, bd2c0, lane);
        __syncthreads();
        build_stage_input<3>(tid, sY, sK, sH, sYS, c4);
        k = feval_core(w, lane, sYS, sAct, sWd0, sWd1, sWd2, bd0c0, bd0c1,
                       bd1c0, bd1c1);
        store_tile_f32(sK + 3 * 2048, L_, 16 * w, k, bd2c0, lane);
        __syncthreads();
        build_stage_input<4>(tid, sY, sK, sH, sYS, c5);
        k = feval_core(w, lane, sYS, sAct, sWd0, sWd1, sWd2, bd0c0, bd0c1,
                       bd1c0, bd1c1);
        store_tile_f32(sK + 4 * 2048, L_, 16 * w, k, bd2c0, lane);
        __syncthreads();
        build_stage_input<5>(tid, sY, sK, sH, sYS, c6);
        // k6: fused y update (wave w owns cols 16w..16w+15)
        k = feval_core(w, lane, sYS, sAct, sWd0, sWd1, sWd2, bd0c0, bd0c1,
                       bd1c0, bd1c1);
        {
          const int n = lane & 15, lg = lane >> 4;
          V8F a8; a8.v = k;
#pragma unroll
          for (int r = 0; r < 8; ++r) {
            const int m = r + 8 * lg;
            const int idx = m * L_ + 16 * w + n;
            const float k6 = a8.e[r] + bd2c0;
            const float comb = B1 * sK[idx] + B3 * sK[2 * 2048 + idx] +
                               B4 * sK[3 * 2048 + idx] +
                               B5 * sK[4 * 2048 + idx] + B6 * k6;
            const float y = sY[idx] + sH[m] * comb;
            sY[idx] = y;
            sYS[idx] = (_Float16)y;
          }
        }
        __syncthreads();
      }
    }

    // ---- GRU: latent = GRUCell(acs[:,t], latent) ---------------------------
    {
      float* grz = sK;                    // 16 x 256 (r | z), f32
      float* gin = sK + 16 * 256;        // 16 x 128, x-part of n (+bih_n)
      float* ghn = gin + 16 * 128;       // 16 x 128, h-part of n
      const v16h ax = load_a_acs(acs, m0, t, lane);
      // r gate: cols [0,128)
      {
        v8f acc = wmma32(ax, load_b_gf32_kpad(Wih, AC_, 16 * w, lane, AC_),
                         v8f{});
#pragma unroll
        for (int kb = 0; kb < L_; kb += 32)
          acc = wmma32(load_a_lds(sYS, L_, kb, lane),
                       load_b_gf32(Whh, L_, 16 * w, kb, lane), acc);
        store_tile_f32(grz, 256, 16 * w, acc, bihR, lane);
      }
      // z gate: cols [128,256)
      {
        v8f acc = wmma32(
            ax, load_b_gf32_kpad(Wih, AC_, L_ + 16 * w, lane, AC_), v8f{});
#pragma unroll
        for (int kb = 0; kb < L_; kb += 32)
          acc = wmma32(load_a_lds(sYS, L_, kb, lane),
                       load_b_gf32(Whh, L_, L_ + 16 * w, kb, lane), acc);
        store_tile_f32(grz, 256, 128 + 16 * w, acc, bihZ, lane);
      }
      // n gate: keep x-part and h-part separate (cols [256,384))
      {
        v8f accI = wmma32(
            ax, load_b_gf32_kpad(Wih, AC_, 2 * L_ + 16 * w, lane, AC_), v8f{});
        v8f accH = {};
#pragma unroll
        for (int kb = 0; kb < L_; kb += 32)
          accH = wmma32(load_a_lds(sYS, L_, kb, lane),
                        load_b_gf32(Whh, L_, 2 * L_ + 16 * w, kb, lane), accH);
        store_tile_f32(gin, L_, 16 * w, accI, bihN, lane);
        store_tile_f32(ghn, L_, 16 * w, accH, 0.f, lane);
      }
      __syncthreads();
      // elementwise gate math: 8 contiguous lanes per thread, vector LDS ops
      {
        const int i0 = tid * 8;
        const int m = i0 >> 7, l = i0 & (L_ - 1);
        V8F gr; gr.v = ld8_f32(grz + m * 256 + l);
        V8F gz; gz.v = ld8_f32(grz + m * 256 + 128 + l);
        V8F gi; gi.v = ld8_f32(gin + i0);
        V8F gh; gh.v = ld8_f32(ghn + i0);
        V8F yv; yv.v = ld8_f32(sY + i0);
        float bnl[8];
        *(float4*)&bnl[0] = *(const float4*)(bn + l);
        *(float4*)&bnl[4] = *(const float4*)(bn + l + 4);
        float yo[8];
#pragma unroll
        for (int e = 0; e < 8; ++e) {
          const float r = sigmoidf_(gr.e[e]);
          const float z = sigmoidf_(gz.e[e]);
          const float nn = tanhf(gi.e[e] + r * (gh.e[e] + bnl[e]));
          yo[e] = (1.f - z) * nn + z * yv.e[e];
        }
        *(float4*)(sY + i0) = *(const float4*)&yo[0];
        *(float4*)(sY + i0 + 4) = *(const float4*)&yo[4];
        st8_f16(sYS + i0, yo);
      }
      __syncthreads();
    }

    // ---- decoder: out[:,t,:] = relu(latent@Wo0^T+bo0) @ Wo1^T + bo1 --------
    {
      v8f d0 = {}, d1 = {};
#pragma unroll
      for (int kb = 0; kb < L_; kb += 32) {
        const v16h af = load_a_lds(sYS, L_, kb, lane);
        d0 = wmma32(af, load_b_gf32(Wo0, L_, 16 * w, kb, lane), d0);
        d1 = wmma32(af, load_b_gf32(Wo0, L_, 16 * (w + 8), kb, lane), d1);
      }
      store_tile_f16(sAct, H_, 16 * w, d0, bo0c0, true, lane);
      store_tile_f16(sAct, H_, 16 * (w + 8), d1, bo0c1, true, lane);
      __syncthreads();
      if (w < 2) {  // N=32 -> 2 tiles; wave-uniform branch keeps EXEC all-ones
        v8f acc = {};
#pragma unroll
        for (int kb = 0; kb < H_; kb += 32)
          acc = wmma32(load_a_lds(sAct, H_, kb, lane),
                       load_b_gf32(Wo1, H_, 16 * w, kb, lane), acc);
        const int n = lane & 15, lg = lane >> 4;
        V8F a8; a8.v = acc;
        const float b = bo1[16 * w + n];
#pragma unroll
        for (int r = 0; r < 8; ++r) {
          const int m = r + 8 * lg;
          out[(size_t)(m0 + m) * T_ * OB_ + (size_t)t * OB_ + 16 * w + n] =
              a8.e[r] + b;
        }
      }
      __syncthreads();
    }
  }
}

extern "C" void kernel_launch(void* const* d_in, const int* in_sizes, int n_in,
                              void* d_out, int out_size, void* d_ws,
                              size_t ws_size, hipStream_t stream) {
  (void)in_sizes; (void)n_in; (void)out_size; (void)d_ws; (void)ws_size;
  const float* ob    = (const float*)d_in[0];
  const float* acs   = (const float*)d_in[1];
  const float* times = (const float*)d_in[2];
  const float* We0 = (const float*)d_in[3];  const float* be0 = (const float*)d_in[4];
  const float* We1 = (const float*)d_in[5];  const float* be1 = (const float*)d_in[6];
  const float* Wd0 = (const float*)d_in[7];  const float* bd0 = (const float*)d_in[8];
  const float* Wd1 = (const float*)d_in[9];  const float* bd1 = (const float*)d_in[10];
  const float* Wd2 = (const float*)d_in[11]; const float* bd2 = (const float*)d_in[12];
  const float* Wo0 = (const float*)d_in[13]; const float* bo0 = (const float*)d_in[14];
  const float* Wo1 = (const float*)d_in[15]; const float* bo1 = (const float*)d_in[16];
  const float* Wih = (const float*)d_in[17]; const float* Whh = (const float*)d_in[18];
  const float* bih = (const float*)d_in[19]; const float* bn  = (const float*)d_in[20];
  float* out = (float*)d_out;

  hipLaunchKernelGGL(ode_rnn_kernel, dim3(B_ / 16), dim3(BDIM), 0, stream, ob,
                     acs, times, We0, be0, We1, be1, Wd0, bd0, Wd1, bd1, Wd2,
                     bd2, Wo0, bo0, Wo1, bo1, Wih, Whh, bih, bn, out);
}
